// ETP_33337536151697
// MI455X (gfx1250) — matrix operand order repaired and checked by reading, not compiled, not run on hardware
//
#include <hip/hip_runtime.h>

// ---------------------------------------------------------------------------
// Sinkhorn OT (FASTopic ETP) for MI455X / gfx1250, wave32, WMMA f16.
//   Phase A: prep (f16 convert + row sq-norms), GEMM via v_wmma_f32_16x16x32_f16
//            -> M (fp32, parked in d_out[1:]) and K16 (f16, L2-resident in ws)
//   Phase B: 100 unrolled Sinkhorn iterations on f16 K (bandwidth phase),
//            device-side convergence flag at it=1 and it=51 (reference exact)
//   Phase C: transp = u * exp(-alpha*M) * v in fp32, deterministic loss sum.
// ---------------------------------------------------------------------------

#define N_TOPICS 1024
#define VOCAB    50000
#define DIMS     256
#define ALPHA    0.05f
#define EPSV     1e-16f
#define INV_N    (1.0f / 1024.0f)
#define INV_M    (1.0f / 50000.0f)
#define STOP_THR 0.005f
#define VHALF    25000   // VOCAB/2 (half2 columns)

typedef _Float16 v16h __attribute__((ext_vector_type(16)));
typedef _Float16 v8h  __attribute__((ext_vector_type(8)));
typedef _Float16 v2h  __attribute__((ext_vector_type(2)));
typedef float    v8f  __attribute__((ext_vector_type(8)));

// ---------------- prep: fp32 -> f16 copy + row squared norms ----------------
// one wave per row; lane handles 8 contiguous elements (16B f16 store)
__global__ __launch_bounds__(256)
void prep_rows(const float* __restrict__ src, int rows,
               _Float16* __restrict__ dst16, float* __restrict__ sq) {
  const int lane = threadIdx.x & 31;
  const int wave = threadIdx.x >> 5;
  const int row  = blockIdx.x * 8 + wave;
  if (row >= rows) return;
  const float* s = src + (size_t)row * DIMS + lane * 8;
  float4 f0 = *(const float4*)(s);
  float4 f1 = *(const float4*)(s + 4);
  float ss = f0.x*f0.x + f0.y*f0.y + f0.z*f0.z + f0.w*f0.w
           + f1.x*f1.x + f1.y*f1.y + f1.z*f1.z + f1.w*f1.w;
  v8h h;
  h[0]=(_Float16)f0.x; h[1]=(_Float16)f0.y; h[2]=(_Float16)f0.z; h[3]=(_Float16)f0.w;
  h[4]=(_Float16)f1.x; h[5]=(_Float16)f1.y; h[6]=(_Float16)f1.z; h[7]=(_Float16)f1.w;
  *(v8h*)(dst16 + (size_t)row * DIMS + lane * 8) = h;
  #pragma unroll
  for (int o = 16; o > 0; o >>= 1) ss += __shfl_down(ss, o);
  if (lane == 0) sq[row] = ss;
}

// ---------------- init u, v, flag ----------------
__global__ __launch_bounds__(256)
void init_uv(float* __restrict__ u, float* __restrict__ v, int* __restrict__ flag) {
  int g = blockIdx.x * 256 + threadIdx.x;
  if (g < VOCAB)    v[g] = 0.0f;
  if (g < N_TOPICS) u[g] = INV_N;
  if (g == 0)       *flag = 0;
}

// ---------------- GEMM: M = xsq + ysq - 2 x yT ; K16 = exp(-alpha*M) --------
// 8 waves/block; wave owns a 16x64 strip (1 i-tile x 4 j-tiles), 8 K-chunks.
__global__ __launch_bounds__(256)
void gemm_mk(const _Float16* __restrict__ x16, const _Float16* __restrict__ y16,
             const float* __restrict__ xsq, const float* __restrict__ ysq,
             float* __restrict__ Mout, _Float16* __restrict__ K16) {
  const int lane = threadIdx.x & 31;
  const int wave = threadIdx.x >> 5;
  const int w      = blockIdx.x * 8 + wave;   // strip id
  const int i_tile = w & 63;                  // 64 i-tiles (1024/16)
  const int jstrip = w >> 6;                  // 782 j-strips of 64
  const int i0 = i_tile << 4;
  const int j0 = jstrip << 6;

  const int lrow = lane & 15;
  const int lhi  = lane >> 4;                 // 0 or 1

  v8f acc[4] = {};

  const _Float16* aRow = x16 + (size_t)(i0 + lrow) * DIMS;
  int bj[4];
  #pragma unroll
  for (int t = 0; t < 4; ++t) {
    int jj = j0 + t * 16 + lrow;
    bj[t] = (jj < VOCAB) ? jj : (VOCAB - 1);  // clamp keeps EXEC uniform
  }

  #pragma unroll
  for (int kc = 0; kc < 8; ++kc) {
    const int kb = kc * 32;
    v16h a;
    {
      v8h lo = *(const v8h*)(aRow + kb + lhi * 8);
      v8h hi = *(const v8h*)(aRow + kb + lhi * 8 + 16);
      #pragma unroll
      for (int e = 0; e < 8; ++e) { a[e] = lo[e]; a[8 + e] = hi[e]; }
    }
    #pragma unroll
    for (int t = 0; t < 4; ++t) {
      const _Float16* bp = y16 + (size_t)bj[t] * DIMS + kb + lhi * 8;
      v8h b0 = *(const v8h*)(bp);
      v8h b1 = *(const v8h*)(bp + 16);
      v16h b;
      #pragma unroll
      for (int e = 0; e < 8; ++e) { b[e] = b0[e]; b[8 + e] = b1[e]; }
      acc[t] = __builtin_amdgcn_wmma_f32_16x16x32_f16(
          false, a, false, b, (short)0, acc[t], false, false);
    }
  }

  #pragma unroll
  for (int t = 0; t < 4; ++t) {
    const int jb = j0 + t * 16;
    if (jb >= VOCAB) continue;                // wave-uniform skip
    const int col = jb + lrow;
    const float ys = ysq[col];
    #pragma unroll
    for (int r = 0; r < 8; ++r) {
      const int row = i0 + lhi * 8 + r;       // C/D layout: VGPR r -> M = r + 8*lhi
      const float m = xsq[row] + ys - 2.0f * acc[t][r];
      Mout[(size_t)row * VOCAB + col] = m;
      K16 [(size_t)row * VOCAB + col] = (_Float16)__expf(-ALPHA * m);
    }
  }
}

// ---------------- Sinkhorn: v = b / (K^T u + eps) ----------------
// thread owns a half2 column pair; u staged in LDS (4KB broadcast reads)
__global__ __launch_bounds__(256)
void sink_v(const v2h* __restrict__ K2, const float* __restrict__ u,
            float* __restrict__ v, const int* __restrict__ flag) {
  if (*flag) return;
  __shared__ float su[N_TOPICS];
  #pragma unroll
  for (int k = 0; k < 4; ++k) su[threadIdx.x + 256 * k] = u[threadIdx.x + 256 * k];
  __syncthreads();
  const int c = blockIdx.x * 256 + threadIdx.x;
  if (c >= VHALF) return;
  float ax = 0.0f, ay = 0.0f;
  const v2h* col = K2 + c;
  #pragma unroll 4
  for (int i = 0; i < N_TOPICS; ++i) {
    v2h k = col[(size_t)i * VHALF];
    __builtin_prefetch((const void*)(col + (size_t)(i + 16) * VHALF), 0, 0);
    float uu = su[i];
    ax += uu * (float)k[0];
    ay += uu * (float)k[1];
  }
  v[2 * c]     = INV_M / (ax + EPSV);
  v[2 * c + 1] = INV_M / (ay + EPSV);
}

// ---------------- Sinkhorn: u = a / (K v + eps) ----------------
// one block per row; deterministic LDS tree reduction
__global__ __launch_bounds__(256)
void sink_u(const v2h* __restrict__ K2, const float* __restrict__ v,
            float* __restrict__ u, const int* __restrict__ flag) {
  if (*flag) return;
  __shared__ float red[256];
  const int row = blockIdx.x;
  const v2h*   kr = K2 + (size_t)row * VHALF;
  const float2* v2 = (const float2*)v;
  float acc = 0.0f;
  for (int c = threadIdx.x; c < VHALF; c += 256) {
    v2h k = kr[c];
    float2 vv = v2[c];
    acc += vv.x * (float)k[0] + vv.y * (float)k[1];
  }
  red[threadIdx.x] = acc;
  __syncthreads();
  #pragma unroll
  for (int s = 128; s > 0; s >>= 1) {
    if (threadIdx.x < s) red[threadIdx.x] += red[threadIdx.x + s];
    __syncthreads();
  }
  if (threadIdx.x == 0) u[row] = INV_N / (red[0] + EPSV);
}

// ---------------- err = sum_j | v_j * (K^T u)_j - 1/m | (per-block partials) -
__global__ __launch_bounds__(256)
void err_partial(const v2h* __restrict__ K2, const float* __restrict__ u,
                 const float* __restrict__ v, float* __restrict__ partials,
                 const int* __restrict__ flag) {
  if (*flag) return;
  __shared__ float su[N_TOPICS];
  __shared__ float red[256];
  #pragma unroll
  for (int k = 0; k < 4; ++k) su[threadIdx.x + 256 * k] = u[threadIdx.x + 256 * k];
  __syncthreads();
  const int c = blockIdx.x * 256 + threadIdx.x;
  float e = 0.0f;
  if (c < VHALF) {
    float ax = 0.0f, ay = 0.0f;
    const v2h* col = K2 + c;
    #pragma unroll 4
    for (int i = 0; i < N_TOPICS; ++i) {
      v2h k = col[(size_t)i * VHALF];
      float uu = su[i];
      ax += uu * (float)k[0];
      ay += uu * (float)k[1];
    }
    e = fabsf(v[2 * c] * ax - INV_M) + fabsf(v[2 * c + 1] * ay - INV_M);
  }
  red[threadIdx.x] = e;
  __syncthreads();
  #pragma unroll
  for (int s = 128; s > 0; s >>= 1) {
    if (threadIdx.x < s) red[threadIdx.x] += red[threadIdx.x + s];
    __syncthreads();
  }
  if (threadIdx.x == 0) partials[blockIdx.x] = red[0];
}

__global__ __launch_bounds__(128)
void err_final(const float* __restrict__ partials, int nparts, int* __restrict__ flag) {
  if (*flag) return;
  __shared__ float red[128];
  float s = 0.0f;
  for (int i = threadIdx.x; i < nparts; i += 128) s += partials[i];
  red[threadIdx.x] = s;
  __syncthreads();
  #pragma unroll
  for (int k = 64; k > 0; k >>= 1) {
    if (threadIdx.x < k) red[threadIdx.x] += red[threadIdx.x + k];
    __syncthreads();
  }
  if (threadIdx.x == 0 && red[0] <= STOP_THR) *flag = 1;
}

// ---------------- transp = u * exp(-alpha*M) * v (fp32), loss partials ------
__global__ __launch_bounds__(256)
void finalize(float* __restrict__ out, const float* __restrict__ u,
              const float* __restrict__ v, float* __restrict__ partials) {
  __shared__ float red[256];
  float* T = out + 1;
  float lsum = 0.0f;
  const size_t base = (size_t)blockIdx.x * 1024;
  #pragma unroll
  for (int k = 0; k < 4; ++k) {
    size_t idx = base + threadIdx.x + 256 * k;
    int i = (int)(idx / VOCAB);
    int j = (int)(idx - (size_t)i * VOCAB);
    float m = T[idx];
    float t = u[i] * __expf(-ALPHA * m) * v[j];
    T[idx] = t;
    lsum += t * m;
  }
  red[threadIdx.x] = lsum;
  __syncthreads();
  #pragma unroll
  for (int s = 128; s > 0; s >>= 1) {
    if (threadIdx.x < s) red[threadIdx.x] += red[threadIdx.x + s];
    __syncthreads();
  }
  if (threadIdx.x == 0) partials[blockIdx.x] = red[0];
}

__global__ __launch_bounds__(256)
void loss_final(const float* __restrict__ partials, float* __restrict__ out) {
  __shared__ float red[256];
  float s = 0.0f;
  for (int i = threadIdx.x; i < VOCAB; i += 256) s += partials[i];
  red[threadIdx.x] = s;
  __syncthreads();
  #pragma unroll
  for (int k = 128; k > 0; k >>= 1) {
    if (threadIdx.x < k) red[threadIdx.x] += red[threadIdx.x + k];
    __syncthreads();
  }
  if (threadIdx.x == 0) out[0] = red[0];
}

// ---------------------------------------------------------------------------
extern "C" void kernel_launch(void* const* d_in, const int* in_sizes, int n_in,
                              void* d_out, int out_size, void* d_ws, size_t ws_size,
                              hipStream_t stream) {
  const float* x = (const float*)d_in[0];   // [1024, 256]
  const float* y = (const float*)d_in[1];   // [50000, 256]
  float* out = (float*)d_out;               // [0]=loss, [1:]=transp fp32

  // workspace carving (~129 MB): K16 is the L2-resident hot buffer
  char* ws = (char*)d_ws;
  size_t off = 0;
  auto carve = [&](size_t bytes) -> void* {
    void* p = ws + off;
    off = (off + bytes + 255) & ~(size_t)255;
    return p;
  };
  _Float16* K16 = (_Float16*)carve((size_t)N_TOPICS * VOCAB * sizeof(_Float16));
  _Float16* x16 = (_Float16*)carve((size_t)N_TOPICS * DIMS * sizeof(_Float16));
  _Float16* y16 = (_Float16*)carve((size_t)VOCAB * DIMS * sizeof(_Float16));
  float* xsq      = (float*)carve(N_TOPICS * sizeof(float));
  float* ysq      = (float*)carve(VOCAB * sizeof(float));
  float* u        = (float*)carve(N_TOPICS * sizeof(float));
  float* v        = (float*)carve(VOCAB * sizeof(float));
  float* partials = (float*)carve(VOCAB * sizeof(float));
  int*   flag     = (int*)carve(64);

  const v2h* K2 = (const v2h*)K16;

  // Phase A: prep + WMMA GEMM (M -> out+1, K16 -> ws)
  prep_rows<<<N_TOPICS / 8, 256, 0, stream>>>(x, N_TOPICS, x16, xsq);
  prep_rows<<<VOCAB / 8,    256, 0, stream>>>(y, VOCAB,    y16, ysq);
  init_uv<<<(VOCAB + 255) / 256, 256, 0, stream>>>(u, v, flag);
  // strips: 64 i-tiles x 782 j-strips(64 wide) = 50048 waves / 8 per block
  gemm_mk<<<(64 * 782) / 8, 256, 0, stream>>>(x16, y16, xsq, ysq, out + 1, K16);

  // Phase B: 100 Sinkhorn iterations, err gate at cpt=1 and cpt=51
  const int vblocks = (VHALF + 255) / 256;   // 98
  for (int it = 1; it <= 100; ++it) {
    sink_v<<<vblocks, 256, 0, stream>>>(K2, u, v, flag);
    sink_u<<<N_TOPICS, 256, 0, stream>>>(K2, v, u, flag);
    if (it == 1 || it == 51) {
      err_partial<<<vblocks, 256, 0, stream>>>(K2, u, v, partials, flag);
      err_final<<<1, 128, 0, stream>>>(partials, vblocks, flag);
    }
  }

  // Phase C: transport plan + loss (deterministic two-stage reduction)
  finalize<<<VOCAB, 256, 0, stream>>>(out, u, v, partials);   // 50000 blocks x 1024 elems
  loss_final<<<1, 256, 0, stream>>>(partials, out);
}
